// QueryEncoder_52931176956201
// MI455X (gfx1250) — matrix operand order repaired and checked by reading, not compiled, not run on hardware
//
#include <hip/hip_runtime.h>
#include <hip/hip_bf16.h>
#include <stdint.h>

// Problem constants (from the reference).
#define BSZ 32768
#define TD  1024
#define HD  512
#define KC  8192

typedef __attribute__((ext_vector_type(16))) __bf16 v16bf;
typedef __attribute__((ext_vector_type(8)))  float  v8f;
typedef int vi4 __attribute__((vector_size(16)));
typedef int vi2 __attribute__((vector_size(8)));

union Frag16 { uint32_t u[8]; uint4 q[2]; v16bf v; };

__device__ __forceinline__ uint32_t pack_hi2(float x0, float x1) {
  // bf16(x0) in low 16, bf16(x1) in high 16 (truncation split)
  return (__float_as_uint(x0) >> 16) | (__float_as_uint(x1) & 0xFFFF0000u);
}
__device__ __forceinline__ float trunc_hi(float x) {
  return __uint_as_float(__float_as_uint(x) & 0xFFFF0000u);
}

// ---- CDNA5 async global->LDS copies (ASYNCcnt path), with safe fallback ----
#if __has_builtin(__builtin_amdgcn_global_load_async_to_lds_b128) && \
    __has_builtin(__builtin_amdgcn_global_load_async_to_lds_b64)
#define ASYNC_LDS 1
#else
#define ASYNC_LDS 0
#endif

__device__ __forceinline__ void copy_g2l_b128(const void* g, void* l) {
#if ASYNC_LDS
  __builtin_amdgcn_global_load_async_to_lds_b128(
      (__attribute__((address_space(1))) vi4*)g,
      (__attribute__((address_space(3))) vi4*)l, 0, 0);
#else
  *(uint4*)l = *(const uint4*)g;
#endif
}
__device__ __forceinline__ void copy_g2l_b64(const void* g, void* l) {
#if ASYNC_LDS
  __builtin_amdgcn_global_load_async_to_lds_b64(
      (__attribute__((address_space(1))) vi2*)g,
      (__attribute__((address_space(3))) vi2*)l, 0, 0);
#else
  *(uint2*)l = *(const uint2*)g;
#endif
}
__device__ __forceinline__ void async_wait0() {
#if ASYNC_LDS
#if __has_builtin(__builtin_amdgcn_s_wait_asynccnt)
  __builtin_amdgcn_s_wait_asynccnt(0);
#else
  asm volatile("s_wait_asynccnt 0x0" ::: "memory");
#endif
#endif
}

// ---------------------------------------------------------------------------
// Kernel 1: split W (fp32 -> bf16 hi/lo). Exactly HD*TD threads.
// ---------------------------------------------------------------------------
__global__ void split_w_kernel(const float* __restrict__ W,
                               unsigned short* __restrict__ wHi,
                               unsigned short* __restrict__ wLo) {
  int i = blockIdx.x * blockDim.x + threadIdx.x;
  float x = W[i];
  uint32_t xb = __float_as_uint(x);
  wHi[i] = (unsigned short)(xb >> 16);
  float lo = x - __uint_as_float(xb & 0xFFFF0000u);
  wLo[i] = (unsigned short)(__float_as_uint(lo) >> 16);
}

// ---------------------------------------------------------------------------
// Kernel 2: split codebook + row norms. One block (256 thr) per codebook row.
// ---------------------------------------------------------------------------
__global__ void split_c_kernel(const float* __restrict__ C,
                               unsigned short* __restrict__ cHi,
                               unsigned short* __restrict__ cLo,
                               float* __restrict__ cSq) {
  __shared__ float red[256];
  int row = blockIdx.x;
  int t = threadIdx.x;
  float ssum = 0.0f;
  for (int e = t; e < HD; e += 256) {
    float x = C[(size_t)row * HD + e];
    uint32_t xb = __float_as_uint(x);
    cHi[(size_t)row * HD + e] = (unsigned short)(xb >> 16);
    float lo = x - __uint_as_float(xb & 0xFFFF0000u);
    cLo[(size_t)row * HD + e] = (unsigned short)(__float_as_uint(lo) >> 16);
    ssum += x * x;
  }
  red[t] = ssum;
  __syncthreads();
  for (int s = 128; s > 0; s >>= 1) {
    if (t < s) red[t] += red[t + s];
    __syncthreads();
  }
  if (t == 0) cSq[row] = red[0];
}

// ---------------------------------------------------------------------------
// Kernel 3: h = tokens @ W^T + b, via bf16-split WMMA.
// Block = 512 thr (16 waves): 2 row-tiles (M) x 8 col-tiles (N) of 16x16.
// Grid = (BSZ/32, HD/128). Token tile double-buffered through LDS with
// async global->LDS copies overlapping the WMMAs.
// ---------------------------------------------------------------------------
__global__ __launch_bounds__(512) void proj_kernel(
    const float* __restrict__ tokens,
    const unsigned short* __restrict__ wHi16,
    const unsigned short* __restrict__ wLo16,
    const float* __restrict__ bias,
    unsigned short* __restrict__ hHi,
    unsigned short* __restrict__ hLo) {
  __shared__ float tk[2][32 * 34];  // double buffer: 32 rows x 32 k, stride 34
  const uint32_t* wHi = (const uint32_t*)wHi16;
  const uint32_t* wLo = (const uint32_t*)wLo16;

  int tid = threadIdx.x;
  int lane = tid & 31, wave = tid >> 5;
  int ln = lane & 15, half = lane >> 4;
  int wm = wave >> 3, wn = wave & 7;
  int m0 = blockIdx.x * 32;
  int n = blockIdx.y * 128 + wn * 16 + ln;
  int ldRow = tid >> 4, ldCol = (tid & 15) * 2;

  const float* gsrc = &tokens[(size_t)(m0 + ldRow) * TD + ldCol];
  float* ldst0 = &tk[0][ldRow * 34 + ldCol];
  float* ldst1 = &tk[1][ldRow * 34 + ldCol];

  copy_g2l_b64(gsrc, ldst0);  // prologue: fill buffer 0 (kt = 0)

  v8f acc = {};
  for (int kt = 0; kt < TD; kt += 32) {
    int buf = (kt >> 5) & 1;
    async_wait0();
    __syncthreads();
    if (kt + 32 < TD)  // issue next tile into the other buffer
      copy_g2l_b64(gsrc + kt + 32, buf ? ldst0 : ldst1);

    Frag16 aHi, aLo, bHi, bLo;
    const float* tkb = &tk[buf][(16 * wm + ln) * 34];
#pragma unroll
    for (int v = 0; v < 8; ++v) {
      // A fragment (16x32 bf16): lane = M, K = 2v + 16*half
      int kk = 2 * v + 16 * half;
      float2 x = *(const float2*)&tkb[kk];
      aHi.u[v] = pack_hi2(x.x, x.y);
      float l0 = x.x - trunc_hi(x.x);
      float l1 = x.y - trunc_hi(x.y);
      aLo.u[v] = pack_hi2(l0, l1);
    }
    // B fragment (32x16 bf16): lane = N, K = 16*half + 2v  (B[k,n] = W[n,k]).
    // Packed-pair u32 indices are consecutive -> two b128 loads each.
    size_t bidx = (size_t)n * (TD / 2) + (kt >> 1) + 8 * half;
    bHi.q[0] = *(const uint4*)&wHi[bidx];
    bHi.q[1] = *(const uint4*)&wHi[bidx + 4];
    bLo.q[0] = *(const uint4*)&wLo[bidx];
    bLo.q[1] = *(const uint4*)&wLo[bidx + 4];

    acc = __builtin_amdgcn_wmma_f32_16x16x32_bf16(false, aHi.v, false, bHi.v,
                                                  (short)0, acc, false, false);
    acc = __builtin_amdgcn_wmma_f32_16x16x32_bf16(false, aHi.v, false, bLo.v,
                                                  (short)0, acc, false, false);
    acc = __builtin_amdgcn_wmma_f32_16x16x32_bf16(false, aLo.v, false, bHi.v,
                                                  (short)0, acc, false, false);
  }

  float bval = bias[n];
#pragma unroll
  for (int r = 0; r < 8; ++r) {
    float x = acc[r] + bval;
    int row = m0 + 16 * wm + r + 8 * half;  // C/D layout: M = r + 8*half
    uint32_t xb = __float_as_uint(x);
    hHi[(size_t)row * HD + n] = (unsigned short)(xb >> 16);
    float lo = x - __uint_as_float(xb & 0xFFFF0000u);
    hLo[(size_t)row * HD + n] = (unsigned short)(__float_as_uint(lo) >> 16);
  }
}

// ---------------------------------------------------------------------------
// Kernel 4: argmin_j ( cSq[j] - 2 * h . c_j )   (hSq & sqrt are argmin-invariant)
// Block = 512 thr (16 waves): 2 row-tiles x 8 j-tiles; j strides by 128 over K.
// The block's 32 h-rows (hi+lo) are staged once into LDS via async copies.
// LDS row stride = 260 u32 (=> distinct banks across the 16 lanes, 16B aligned).
// Grid = BSZ/32.
// ---------------------------------------------------------------------------
#define HROW 260  // u32 stride per 512-bf16 row (256 data + 4 pad)

__global__ __launch_bounds__(512) void dist_kernel(
    const unsigned short* __restrict__ hHi16,
    const unsigned short* __restrict__ hLo16,
    const unsigned short* __restrict__ cHi16,
    const unsigned short* __restrict__ cLo16,
    const float* __restrict__ cSq,
    long long* __restrict__ out) {
  __shared__ uint32_t ldsHi[32 * HROW];
  __shared__ uint32_t ldsLo[32 * HROW];
  __shared__ unsigned long long sbest[32];
  const uint32_t* cHi = (const uint32_t*)cHi16;
  const uint32_t* cLo = (const uint32_t*)cLo16;

  int tid = threadIdx.x;
  int lane = tid & 31, wave = tid >> 5;
  int ln = lane & 15, half = lane >> 4;
  int wm = wave >> 3, wn = wave & 7;
  int m0 = blockIdx.x * 32;

  // Stage h rows m0..m0+31 (hi & lo planes) into LDS: 2 x 2048 16B chunks.
#pragma unroll
  for (int c = 0; c < 4; ++c) {
    int chunk = tid + c * 512;            // 0..2047
    int row = chunk >> 6;                 // 64 chunks (1024B) per row
    int seg = chunk & 63;
    size_t gelem = (size_t)(m0 + row) * HD + seg * 8;  // 8 bf16 per chunk
    copy_g2l_b128(hHi16 + gelem, &ldsHi[row * HROW + seg * 4]);
    copy_g2l_b128(hLo16 + gelem, &ldsLo[row * HROW + seg * 4]);
  }
  if (tid < 32) sbest[tid] = ~0ULL;
  async_wait0();
  __syncthreads();

  unsigned long long best[8];
#pragma unroll
  for (int r = 0; r < 8; ++r) best[r] = ~0ULL;

  int rowL = 16 * wm + ln;  // A layout: lane = M
  for (int it = 0; it < KC / 128; ++it) {
    int j = it * 128 + wn * 16 + ln;  // this lane's output column (code index)
    float csq = cSq[j];
    v8f acc = {};
    for (int kt = 0; kt < HD; kt += 32) {
      Frag16 aHi, aLo, bHi, bLo;
      // A fragment from LDS: u32 indices base+0..7 (two ds_load_b128).
      int abase = rowL * HROW + (kt >> 1) + 8 * half;
      aHi.q[0] = *(const uint4*)&ldsHi[abase];
      aHi.q[1] = *(const uint4*)&ldsHi[abase + 4];
      aLo.q[0] = *(const uint4*)&ldsLo[abase];
      aLo.q[1] = *(const uint4*)&ldsLo[abase + 4];
      // B fragment (codebook, B[k,j] = codebook[j,k]): two b128 loads each.
      size_t bidx = (size_t)j * (HD / 2) + (kt >> 1) + 8 * half;
      bHi.q[0] = *(const uint4*)&cHi[bidx];
      bHi.q[1] = *(const uint4*)&cHi[bidx + 4];
      bLo.q[0] = *(const uint4*)&cLo[bidx];
      bLo.q[1] = *(const uint4*)&cLo[bidx + 4];

      acc = __builtin_amdgcn_wmma_f32_16x16x32_bf16(false, aHi.v, false, bHi.v,
                                                    (short)0, acc, false, false);
      acc = __builtin_amdgcn_wmma_f32_16x16x32_bf16(false, aHi.v, false, bLo.v,
                                                    (short)0, acc, false, false);
      acc = __builtin_amdgcn_wmma_f32_16x16x32_bf16(false, aLo.v, false, bHi.v,
                                                    (short)0, acc, false, false);
    }
#pragma unroll
    for (int r = 0; r < 8; ++r) {
      float d2 = csq - 2.0f * acc[r];
      uint32_t b = __float_as_uint(d2);
      b = (b & 0x80000000u) ? ~b : (b | 0x80000000u);  // total-order key
      unsigned long long key = ((unsigned long long)b << 32) | (unsigned)j;
      best[r] = key < best[r] ? key : best[r];
    }
  }

  // Reduce across the 16 lanes (columns) of each half-wave.
#pragma unroll
  for (int r = 0; r < 8; ++r) {
#pragma unroll
    for (int off = 1; off <= 8; off <<= 1) {
      unsigned long long o = __shfl_xor(best[r], off, 32);
      best[r] = o < best[r] ? o : best[r];
    }
  }
  if (ln == 0) {  // lanes 0 and 16 of each wave
    int rbase = 16 * wm + 8 * half;
#pragma unroll
    for (int r = 0; r < 8; ++r) atomicMin(&sbest[rbase + r], best[r]);
  }
  __syncthreads();
  if (tid < 32) out[m0 + tid] = (long long)(sbest[tid] & 0xFFFFFFFFu);
}

// ---------------------------------------------------------------------------
extern "C" void kernel_launch(void* const* d_in, const int* in_sizes, int n_in,
                              void* d_out, int out_size, void* d_ws, size_t ws_size,
                              hipStream_t stream) {
  const float* tokens   = (const float*)d_in[0];
  const float* W        = (const float*)d_in[1];
  const float* bias     = (const float*)d_in[2];
  const float* codebook = (const float*)d_in[3];

  // Workspace carve-up (~82 MB total)
  unsigned short* wHi = (unsigned short*)d_ws;
  unsigned short* wLo = wHi + (size_t)HD * TD;
  unsigned short* cHi = wLo + (size_t)HD * TD;
  unsigned short* cLo = cHi + (size_t)KC * HD;
  unsigned short* hHi = cLo + (size_t)KC * HD;
  unsigned short* hLo = hHi + (size_t)BSZ * HD;
  float*          cSq = (float*)(hLo + (size_t)BSZ * HD);
  long long* out = (long long*)d_out;

  split_w_kernel<<<(HD * TD) / 512, 512, 0, stream>>>(W, wHi, wLo);
  split_c_kernel<<<KC, 256, 0, stream>>>(codebook, cHi, cLo, cSq);
  proj_kernel<<<dim3(BSZ / 32, HD / 128), 512, 0, stream>>>(tokens, wHi, wLo,
                                                            bias, hHi, hLo);
  dist_kernel<<<BSZ / 32, 512, 0, stream>>>(hHi, hLo, cHi, cLo, cSq, out);
}
